// GCNPlusConv_27419071218304
// MI455X (gfx1250) — matrix operand order repaired and checked by reading, not compiled, not run on hardware
//
#include <hip/hip_runtime.h>
#include <hip/hip_bf16.h>

typedef __attribute__((ext_vector_type(16))) _Float16 v16h;
typedef __attribute__((ext_vector_type(8)))  float    v8f;

#define D_DIM 128
#define O_DIM 64
#define STEPS 10

// ---------------- degree / normalization ----------------

__global__ void k_init_deg(float* __restrict__ deg, int N) {
  int i = blockIdx.x * blockDim.x + threadIdx.x;
  if (i < N) deg[i] = 1.0f;  // self loop
}

__global__ void k_deg(const long long* __restrict__ ei, float* __restrict__ deg, int E) {
  int e = blockIdx.x * blockDim.x + threadIdx.x;
  if (e < E) atomicAdd(&deg[(int)ei[(size_t)E + e]], 1.0f);  // col = target index
}

__global__ void k_dinv(float* __restrict__ deg, int N) {
  int i = blockIdx.x * blockDim.x + threadIdx.x;
  if (i < N) deg[i] = rsqrtf(deg[i]);  // in place: deg -> deg^-0.5
}

// ---------------- per-channel softmax over the 10 hop weights ----------------

__global__ void k_softmax(const float* __restrict__ t, float* __restrict__ tn) {
  int d = threadIdx.x;
  if (d < D_DIM) {
    float m = t[d];
#pragma unroll
    for (int k = 1; k < STEPS; ++k) m = fmaxf(m, t[k * D_DIM + d]);
    float ex[STEPS], s = 0.0f;
#pragma unroll
    for (int k = 0; k < STEPS; ++k) { ex[k] = __expf(t[k * D_DIM + d] - m); s += ex[k]; }
    float inv = 1.0f / s;
#pragma unroll
    for (int k = 0; k < STEPS; ++k) tn[k * D_DIM + d] = ex[k] * inv;
  }
}

// ---------------- hop: self-loop init + edge scatter ----------------

// hn[i,d] = dinv[i]^2 * h[i,d]   (fully initializes hn)
__global__ void k_selfinit(const float* __restrict__ h, const float* __restrict__ dinv,
                           float* __restrict__ hn, int N) {
  int idx = blockIdx.x * blockDim.x + threadIdx.x;
  if (idx < N * D_DIM) {
    int i = idx >> 7;
    float di = dinv[i];
    hn[idx] = di * di * h[idx];
  }
}

// hn[col,d] += dinv[row]*dinv[col] * h[row,d] ; 2 edges per 256-thread block
__global__ void k_scatter(const long long* __restrict__ ei, const float* __restrict__ h,
                          const float* __restrict__ dinv, float* __restrict__ hn, int E) {
  int e = blockIdx.x * 2 + (threadIdx.x >> 7);
  int d = threadIdx.x & 127;
  if (e < E) {
    int r = (int)ei[e];
    int c = (int)ei[(size_t)E + e];
    float nrm = dinv[r] * dinv[c];
    atomicAdd(&hn[c * D_DIM + d], nrm * h[r * D_DIM + d]);
  }
}

// ---------------- fused (t_k ⊙ h_k) @ W^T accumulation via WMMA ----------------
// block = 128 threads = 4 waves; wave w handles output-column tile w (16 cols),
// block handles 16 rows. A-operand is scaled by tn_k per K-column on load.
__global__ void k_gemm_wmma(const float* __restrict__ A, const float* __restrict__ W,
                            const float* __restrict__ bias, const float* __restrict__ tnk,
                            float* __restrict__ out, int N, int beta) {
  const int lane = threadIdx.x & 31;
  const int wave = threadIdx.x >> 5;      // o-tile 0..3
  const int half = lane >> 4;             // K-half selector per ISA layout
  const int m    = lane & 15;             // A row within tile / B column within tile
  const int rowbase = blockIdx.x * 16;
  const int row  = rowbase + m;
  const int rA   = (row < N) ? row : (N - 1);
  const int o    = wave * 16 + m;         // W row (output channel) for this lane

  v8f acc = {0.f, 0.f, 0.f, 0.f, 0.f, 0.f, 0.f, 0.f};

#pragma unroll
  for (int kt = 0; kt < 4; ++kt) {        // K = 128 in chunks of 32
    const int kbase = kt * 32;
    // A (16x32 f16): lane holds row m; elements 0..7 -> K = 8*half+e,
    // elements 8..15 -> K = 16 + 8*half + (e-8)   (relative to kbase)
    const float* Ap = A + (size_t)rA * D_DIM + kbase + 8 * half;
    const float* Tp = tnk + kbase + 8 * half;
    v16h a;
#pragma unroll
    for (int e = 0; e < 8; ++e) a[e]     = (_Float16)(Ap[e]      * Tp[e]);
#pragma unroll
    for (int e = 0; e < 8; ++e) a[8 + e] = (_Float16)(Ap[16 + e] * Tp[16 + e]);
    // B (32x16 f16): lane holds column o; elements 0..7 -> K = 8*half+e,
    // elements 8..15 -> K = 16 + 8*half + (e-8)   (relative to kbase)
    const float* Wp = W + (size_t)o * D_DIM + kbase + 8 * half;
    v16h bm;
#pragma unroll
    for (int e = 0; e < 8; ++e) { bm[e] = (_Float16)Wp[e]; bm[8 + e] = (_Float16)Wp[16 + e]; }

    acc = __builtin_amdgcn_wmma_f32_16x16x32_f16(false, a, false, bm,
                                                 (short)0, acc, false, false);
  }

  // C/D layout: VGPR r -> M = r + 8*half, N = lane&15
#pragma unroll
  for (int r = 0; r < 8; ++r) {
    int rr = rowbase + r + 8 * half;
    if (rr < N) {
      size_t idx = (size_t)rr * O_DIM + o;
      float v = acc[r];
      out[idx] = beta ? (out[idx] + v) : (v + bias[o]);
    }
  }
}

// ---------------- host orchestration ----------------

extern "C" void kernel_launch(void* const* d_in, const int* in_sizes, int n_in,
                              void* d_out, int out_size, void* d_ws, size_t ws_size,
                              hipStream_t stream) {
  const float*     x    = (const float*)d_in[0];
  const long long* ei   = (const long long*)d_in[1];  // int64 [2, E]
  const float*     t    = (const float*)d_in[2];
  const float*     W    = (const float*)d_in[3];
  const float*     bias = (const float*)d_in[4];
  float*           out  = (float*)d_out;

  const int N = in_sizes[0] / D_DIM;
  const int E = in_sizes[1] / 2;

  float* bufA = (float*)d_ws;                       // N*D
  float* bufB = bufA + (size_t)N * D_DIM;           // N*D
  float* dinv = bufB + (size_t)N * D_DIM;           // N   (deg -> dinv in place)
  float* tn   = dinv + N;                           // STEPS*D

  k_init_deg<<<(N + 255) / 256, 256, 0, stream>>>(dinv, N);
  k_deg<<<(E + 255) / 256, 256, 0, stream>>>(ei, dinv, E);
  k_dinv<<<(N + 255) / 256, 256, 0, stream>>>(dinv, N);
  k_softmax<<<1, 128, 0, stream>>>(t, tn);

  const int gGemm = (N + 15) / 16;
  // hop 0: out = (tn0 ⊙ x) @ W^T + b
  k_gemm_wmma<<<gGemm, 128, 0, stream>>>(x, W, bias, tn, out, N, 0);

  const float* hprev = x;
  float* bufs[2] = {bufA, bufB};
  int cur = 0;
  const int total = N * D_DIM;
  for (int k = 1; k < STEPS; ++k) {
    float* hn = bufs[cur];
    k_selfinit<<<(total + 255) / 256, 256, 0, stream>>>(hprev, dinv, hn, N);
    k_scatter<<<(E + 1) / 2, 256, 0, stream>>>(ei, hprev, dinv, hn, E);
    k_gemm_wmma<<<gGemm, 128, 0, stream>>>(hn, W, bias, tn + k * D_DIM, out, N, 1);
    hprev = hn;
    cur ^= 1;
  }
}
